// Head_47510928228486
// MI455X (gfx1250) — matrix operand order repaired and checked by reading, not compiled, not run on hardware
//
#include <hip/hip_runtime.h>

// ---------------------------------------------------------------------------
// Fused causal attention head for MI455X (gfx1250, wave32, WMMA).
// B=256, T=512, C=HS=64.
// Kernel 1: QKV projection via v_wmma_f32_16x16x32_f16 (wq pre-scaled 0.125).
// Kernel 2: flash-attention; K/V tiles double-buffered in LDS via
//           global_load_async_to_lds_b128 (ASYNCcnt), S/P tiles LDS-only.
// ---------------------------------------------------------------------------

typedef __attribute__((ext_vector_type(16))) _Float16 v16h;
typedef __attribute__((ext_vector_type(8)))  _Float16 v8h;
typedef __attribute__((ext_vector_type(8)))  float    v8f;

#define NB   256
#define NT   512
#define ND   64
#define BT   (NB * NT)

union VA {
    v16h v;
    v8h  h[2];
    _Float16 e[16];
};
union V8 {
    v8h v;
    _Float16 e[8];
};

static __device__ __forceinline__ v8f wmma_f16(v16h a, v16h b, v8f c) {
    // D = A(16x32 f16) * B(32x16 f16) + C(16x16 f32)
    return __builtin_amdgcn_wmma_f32_16x16x32_f16(
        /*neg_a=*/false, a, /*neg_b=*/false, b,
        /*c_mod=*/(short)0, c, /*reuse_a=*/false, /*reuse_b=*/false);
}

// ---- CDNA5 async global->LDS copy (ASYNCcnt tracked) ----------------------
// VGLOBAL encoding: VDST = per-lane LDS byte address, VADDR = 64-bit global.
// Low 32 bits of a generic LDS pointer are exactly the hardware LDS offset.
static __device__ __forceinline__ void async_load_b128(void* lds_ptr, const void* gptr) {
    unsigned lds = (unsigned)(unsigned long long)lds_ptr;
    asm volatile("global_load_async_to_lds_b128 %0, %1, off"
                 :: "v"(lds), "v"(gptr) : "memory");
}
static __device__ __forceinline__ void wait_async0() {
    asm volatile("s_wait_asynccnt 0x0" ::: "memory");
}

// A-fragment (16x32, 16-bit): lane m = lane&15, hi = lane>>4.
// element e: K = (e<8 ? 8*hi+e : 16 + 8*hi + (e-8))
static __device__ __forceinline__ v16h loadA_f32(const float* rowp, int k0, int hi) {
    VA u;
#pragma unroll
    for (int i = 0; i < 8; ++i) u.e[i]     = (_Float16)rowp[k0 + 8 * hi + i];
#pragma unroll
    for (int i = 0; i < 8; ++i) u.e[8 + i] = (_Float16)rowp[k0 + 16 + 8 * hi + i];
    return u.v;
}

static __device__ __forceinline__ v16h loadA_f16(const _Float16* rowp, int k0, int hi) {
    VA u;
    u.h[0] = *(const v8h*)(rowp + k0 + 8 * hi);
    u.h[1] = *(const v8h*)(rowp + k0 + 16 + 8 * hi);
    return u.v;
}

// B-fragment (32x16, 16-bit) from an N-major tile (stride 64 halfs):
// lane holds column N = n0 + (lane&15); element e: K = k0 + (e<8 ? 8*hi+e : 16 + 8*hi + (e-8)).
static __device__ __forceinline__ v16h loadB_nmajor(const _Float16* tile, int n0, int k0, int lane) {
    const _Float16* p = tile + (n0 + (lane & 15)) * 64 + k0 + 8 * (lane >> 4);
    VA u;
    u.h[0] = *(const v8h*)p;
    u.h[1] = *(const v8h*)(p + 16);
    return u.v;
}

// ---------------------------------------------------------------------------
// Kernel 1: q = x@wq*0.125 (f16), k = x@wk (f16), vT[b][d][t] = (x@wv)^T (f16)
// block = 128 threads (4 waves), each wave computes 16 rows; grid = BT/64.
// ---------------------------------------------------------------------------
__global__ __launch_bounds__(128) void qkv_proj(
    const float* __restrict__ x,
    const float* __restrict__ wk,
    const float* __restrict__ wq,
    const float* __restrict__ wv,
    _Float16* __restrict__ qh,
    _Float16* __restrict__ kh,
    _Float16* __restrict__ vt)
{
    __shared__ _Float16 Wt[3][64 * 64];   // out-major (transposed) f16 weights

    const int tid = threadIdx.x;
    for (int idx = tid; idx < 64 * 64; idx += 128) {
        int o = idx >> 6, i = idx & 63;   // Wt[o*64+i] = W[i][o]
        Wt[0][idx] = (_Float16)(wq[i * 64 + o] * 0.125f);  // fold C^-0.5
        Wt[1][idx] = (_Float16)(wk[i * 64 + o]);
        Wt[2][idx] = (_Float16)(wv[i * 64 + o]);
    }
    __syncthreads();

    const int wave = tid >> 5, lane = tid & 31;
    const int hi = lane >> 4, nlo = lane & 15;
    const int row0 = blockIdx.x * 64 + wave * 16;

    const float* xr = x + (row0 + nlo) * 64;
    const v16h a0 = loadA_f32(xr, 0, hi);
    const v16h a1 = loadA_f32(xr, 32, hi);

    const int bq = row0 >> 9;      // batch
    const int t0 = row0 & 511;     // time offset in batch

#pragma unroll
    for (int w = 0; w < 3; ++w) {
#pragma unroll
        for (int j = 0; j < 4; ++j) {
            v16h b0 = loadB_nmajor(Wt[w], j * 16, 0, lane);
            v16h b1 = loadB_nmajor(Wt[w], j * 16, 32, lane);
            v8f c = {};
            c = wmma_f16(a0, b0, c);
            c = wmma_f16(a1, b1, c);
            const int col = j * 16 + nlo;
            if (w == 0) {
#pragma unroll
                for (int r = 0; r < 8; ++r)
                    qh[(row0 + r + 8 * hi) * 64 + col] = (_Float16)c[r];
            } else if (w == 1) {
#pragma unroll
                for (int r = 0; r < 8; ++r)
                    kh[(row0 + r + 8 * hi) * 64 + col] = (_Float16)c[r];
            } else {
                // transpose on the fly: rows r are contiguous t -> one b128 store
                V8 t;
#pragma unroll
                for (int r = 0; r < 8; ++r) t.e[r] = (_Float16)c[r];
                *(v8h*)(vt + (bq * 64 + col) * 512 + t0 + 8 * hi) = t.v;
            }
        }
    }
}

// ---------------------------------------------------------------------------
// Kernel 2: flash attention. block = 128 threads (4 waves), one batch x 64
// query rows per block; wave w owns query rows q0+16w .. q0+16w+15.
// grid = NB * (NT/64) = 2048. Double-buffered async K/V tiles.
// ---------------------------------------------------------------------------

// issue async global->LDS copies for one 64x64 K tile and one 64x64 V^T tile
static __device__ __forceinline__ void stage_tiles_async(
    _Float16* ksDst, _Float16* vtDst,
    const _Float16* ksrc, const _Float16* vsrc, int tid)
{
#pragma unroll
    for (int it = 0; it < 4; ++it) {          // K tile: flat 8KB copy
        int c = tid + it * 128;
        async_load_b128(ksDst + c * 8, ksrc + c * 8);
    }
#pragma unroll
    for (int it = 0; it < 4; ++it) {          // V^T tile: 64 rows x 128B
        int c = tid + it * 128;
        int d = c >> 3, s0 = (c & 7) * 8;
        async_load_b128(vtDst + d * 64 + s0, vsrc + d * 512 + s0);
    }
}

__global__ __launch_bounds__(128) void attn(
    const _Float16* __restrict__ qh,
    const _Float16* __restrict__ kh,
    const _Float16* __restrict__ vt,
    float* __restrict__ out)
{
    __shared__ _Float16 Ks[2][64 * 64];   // [s][d] row-major key tiles
    __shared__ _Float16 Vts[2][64 * 64];  // [d][s] transposed value tiles
    __shared__ float    Sb[4][16 * 64];   // per-wave S/P staging
    __shared__ float    Arow[4][16];      // per-wave row rescale factors

    const int tid = threadIdx.x, wave = tid >> 5, lane = tid & 31;
    const int hi = lane >> 4, nlo = lane & 15;
    const int b  = blockIdx.x >> 3;
    const int qt = blockIdx.x & 7;
    const int q0 = qt * 64;
    const int qrow0 = q0 + wave * 16;

    const _Float16* kbase = kh + (b * 512) * 64;
    const _Float16* vbase = vt + b * 64 * 512;

    // prologue: kick off async loads of tile 0
    stage_tiles_async(Ks[0], Vts[0], kbase, vbase, tid);

    // Q fragments held in registers for the whole kernel (overlaps with async)
    const _Float16* qrow = qh + (b * 512 + qrow0 + nlo) * 64;
    const v16h aq0 = loadA_f16(qrow, 0, hi);
    const v16h aq1 = loadA_f16(qrow, 32, hi);

    v8f oacc[4] = {{}, {}, {}, {}};
    float m_run = -1e30f, l_run = 0.0f;
    float* Sw = Sb[wave];

    for (int kt = 0; kt <= qt; ++kt) {
        wait_async0();        // this wave's async tile writes have landed in LDS
        __syncthreads();      // ... and everyone else's too
        const int cur = kt & 1;

        if (kt < qt)          // overlap next tile's loads with this tile's math
            stage_tiles_async(Ks[cur ^ 1], Vts[cur ^ 1],
                              kbase + (kt + 1) * 64 * 64,
                              vbase + (kt + 1) * 64, tid);

        const _Float16* Kc = Ks[cur];
        const _Float16* Vc = Vts[cur];

        // ---- S = (Q * 0.125) @ K^T, 16x64 per wave --------------------------
#pragma unroll
        for (int j = 0; j < 4; ++j) {
            v16h bk0 = loadB_nmajor(Kc, j * 16, 0, lane);
            v16h bk1 = loadB_nmajor(Kc, j * 16, 32, lane);
            v8f s = {};
            s = wmma_f16(aq0, bk0, s);
            s = wmma_f16(aq1, bk1, s);
            if (kt == qt) {                       // causal mask on diagonal tile
                const int colr = j * 16 + nlo;    // key index - q0
#pragma unroll
                for (int r = 0; r < 8; ++r)
                    if (colr > wave * 16 + r + 8 * hi) s[r] = -1e30f;
            }
#pragma unroll
            for (int r = 0; r < 8; ++r)
                Sw[(r + 8 * hi) * 64 + j * 16 + nlo] = s[r];
        }

        // ---- online softmax: lane pair (row, row+16) splits the 64 cols -----
        {
            float* rp = Sw + nlo * 64 + 32 * hi;
            float tmax = -1e30f;
#pragma unroll
            for (int i = 0; i < 32; ++i) tmax = fmaxf(tmax, rp[i]);
            tmax = fmaxf(tmax, __shfl_xor(tmax, 16, 32));
            const float newm = fmaxf(m_run, tmax);
            const float al = __expf(m_run - newm);
            float sum = 0.0f;
#pragma unroll
            for (int i = 0; i < 32; ++i) {
                float p = __expf(rp[i] - newm);
                rp[i] = p;
                sum += p;
            }
            sum += __shfl_xor(sum, 16, 32);
            m_run = newm;
            l_run = l_run * al + sum;
            if (hi == 0) Arow[wave][nlo] = al;
        }

        // rescale running O by alpha (row = r + 8*hi in D layout)
#pragma unroll
        for (int j = 0; j < 4; ++j)
#pragma unroll
            for (int r = 0; r < 8; ++r)
                oacc[j][r] *= Arow[wave][r + 8 * hi];

        // rebuild P as f16 A-fragments from LDS
        const float* prow = Sw + nlo * 64;
        const v16h p0 = loadA_f32(prow, 0, hi);
        const v16h p1 = loadA_f32(prow, 32, hi);

        // ---- O += P @ V ------------------------------------------------------
#pragma unroll
        for (int j = 0; j < 4; ++j) {
            v16h bv0 = loadB_nmajor(Vc, j * 16, 0, lane);
            v16h bv1 = loadB_nmajor(Vc, j * 16, 32, lane);
            oacc[j] = wmma_f16(p0, bv0, oacc[j]);
            oacc[j] = wmma_f16(p1, bv1, oacc[j]);
        }
    }

    // ---- final 1/l normalization + store --------------------------------
    if (hi == 0) Arow[wave][nlo] = 1.0f / l_run;
#pragma unroll
    for (int j = 0; j < 4; ++j) {
        const int col = j * 16 + nlo;
#pragma unroll
        for (int r = 0; r < 8; ++r) {
            out[(b * 512 + qrow0 + r + 8 * hi) * 64 + col] =
                oacc[j][r] * Arow[wave][r + 8 * hi];
        }
    }
}

// ---------------------------------------------------------------------------
extern "C" void kernel_launch(void* const* d_in, const int* in_sizes, int n_in,
                              void* d_out, int out_size, void* d_ws, size_t ws_size,
                              hipStream_t stream) {
    const float* x  = (const float*)d_in[0];   // [B,T,C]
    const float* wk = (const float*)d_in[1];   // [C,HS]
    const float* wq = (const float*)d_in[2];   // [C,HS]
    const float* wv = (const float*)d_in[3];   // [C,HS]

    _Float16* qh = (_Float16*)d_ws;                       // [B*T,64] f16 (pre-scaled)
    _Float16* kh = qh + (size_t)BT * ND;                  // [B*T,64] f16
    _Float16* vt = kh + (size_t)BT * ND;                  // [B,64,T] f16 (transposed)

    qkv_proj<<<BT / 64, 128, 0, stream>>>(x, wk, wq, wv, qh, kh, vt);
    attn<<<NB * (NT / 64), 128, 0, stream>>>(qh, kh, vt, (float*)d_out);
}
